// _CrossAttnProcessor_65146063946460
// MI455X (gfx1250) — matrix operand order repaired and checked by reading, not compiled, not run on hardware
//
#include <hip/hip_runtime.h>
#include <hip/hip_bf16.h>
#include <math.h>

typedef __attribute__((ext_vector_type(16))) __bf16 v16bf;
typedef __attribute__((ext_vector_type(8)))  __bf16 v8bf;
typedef __attribute__((ext_vector_type(8)))  float  v8f;

// Pack two f32 -> one dword of 2 bf16 (lowers to v_cvt_pk_bf16_f32).
static __device__ __forceinline__ unsigned pk2(float a, float b) {
  union { __bf16 h[2]; unsigned u; } x;
  x.h[0] = (__bf16)a;
  x.h[1] = (__bf16)b;
  return x.u;
}

// ---------------------------------------------------------------------------
// CDNA5 WMMA bf16 fragment layouts (ISA 7.12.2, wave32):
//  A (16x32): lane = half*16 + m, 8 K-adjacent elems (K%8==0) -> slots
//             8*(K>=16) + 0..7 at lane (Kq>=8)*16 + m  => 16B copy
//  B (32x16): lane = half*16 + n, slot s -> K = 16*(s>=8) + 8*half + (s&7)
//             4 K-adjacent elems (K%4==0) -> contiguous slots => 8B copy
//  C/D (16x16 f32): VGPR r -> row r + 8*half, col = lane&15
// With bf16 sources, tile staging is pure data movement (no VALU).
// ---------------------------------------------------------------------------

// ---- prep: elementwise f32 -> bf16 ----------------------------------------
__global__ __launch_bounds__(256) void convx(const float* __restrict__ X,
                                             __bf16* __restrict__ Y) {
  long i = ((long)blockIdx.x * 256 + threadIdx.x) * 4;
  float4 f = *(const float4*)(X + i);
  uint2 q;
  q.x = pk2(f.x, f.y);
  q.y = pk2(f.z, f.w);
  *(uint2*)(Y + i) = q;
}

// ---- prep: f32 -> bf16 with zero row padding (rows >= Mvalid) -------------
__global__ __launch_bounds__(256) void convpad(const float* __restrict__ X,
                                               __bf16* __restrict__ Y,
                                               int Mvalid, int K) {
  long i = ((long)blockIdx.x * 256 + threadIdx.x) * 4;
  long r = i / K;
  float4 f = {0.f, 0.f, 0.f, 0.f};
  if (r < Mvalid) f = *(const float4*)(X + i);
  uint2 q;
  q.x = pk2(f.x, f.y);
  q.y = pk2(f.z, f.w);
  *(uint2*)(Y + i) = q;
}

// ---- prep: transpose + convert weight W[K][N] f32 -> Wt[N][K] bf16 --------
__global__ __launch_bounds__(256) void wtrans(const float* __restrict__ W,
                                              __bf16* __restrict__ Wt,
                                              int K, int N) {
  __shared__ float tile[32][33];
  const int bx = blockIdx.x * 32;  // n
  const int by = blockIdx.y * 32;  // k
  const int tx = threadIdx.x & 31, ty = threadIdx.x >> 5;  // ty 0..7
#pragma unroll
  for (int i = 0; i < 32; i += 8)
    tile[ty + i][tx] = W[(long)(by + ty + i) * N + bx + tx];
  __syncthreads();
#pragma unroll
  for (int i = 0; i < 32; i += 8)
    Wt[(long)(bx + ty + i) * K + by + tx] = (__bf16)tile[tx][ty + i];
}

// ---------------------------------------------------------------------------
// GEMM, all-bf16 operands: C[M,N] = A[M,K] * Bt[N,K]^T (+bias).
// Block: 256 threads = 8 waves; tile 256(M) x 64(N), K-step 32.
// Each wave owns 2 M-subtiles -> every B fragment feeds 2 WMMAs.
// Double-buffered LDS, one barrier per K-step; staging is raw b128/b64 copies.
// Requires M%256==0, N%64==0, Kt%32==0.
// ---------------------------------------------------------------------------
template <bool OUT_F32>
__global__ __launch_bounds__(256) void gemm_bb(const __bf16* __restrict__ A,
                                               const __bf16* __restrict__ Bt,  // [N][K]
                                               const float* __restrict__ bias,
                                               void* __restrict__ Cp,
                                               int M, int N, int Kt) {
  __shared__ __align__(32) __bf16 Asw[2][16][32][16];  // 2 x 16 KB
  __shared__ __align__(32) __bf16 Bsw[2][4][32][16];   // 2 x  4 KB
  const int t = threadIdx.x;
  const int lane = t & 31;
  const int w = t >> 5;
  const int half = (lane >> 4) & 1;
  const int m = lane & 15;
  const long row0 = (long)blockIdx.y * 256;
  const long col0 = (long)blockIdx.x * 64;

  uint4 ra[4];   // 4 x 8 bf16 (A)
  uint2 rb[2];   // 2 x 4 bf16 (B)

  auto gload = [&](int kt) {
#pragma unroll
    for (int i = 0; i < 4; ++i) {
      int p = t + i * 256;          // 256 rows x 4 k-groups of 8
      int r = p >> 2;
      int kg = (p & 3) * 8;
      ra[i] = *(const uint4*)(A + (row0 + r) * (long)Kt + kt + kg);
    }
#pragma unroll
    for (int i = 0; i < 2; ++i) {   // 64 cols x 8 k-groups of 4
      int p = t + i * 256;
      int c = p >> 3;
      int kg = (p & 7) * 4;
      rb[i] = *(const uint2*)(Bt + (long)(col0 + c) * Kt + kt + kg);
    }
  };
  auto sstore = [&](int buf) {
#pragma unroll
    for (int i = 0; i < 4; ++i) {
      int p = t + i * 256;
      int r = p >> 2;
      int kg = (p & 3) * 8;         // 0,8,16,24
      int wt = r >> 4, mm = r & 15;
      int hi16 = (kg >> 4) & 1;     // slot bank
      int hf = (kg >> 3) & 1;       // lane bank ((kg&15)>=8)
      *(uint4*)(&Asw[buf][wt][hf * 16 + mm][8 * hi16]) = ra[i];
    }
#pragma unroll
    for (int i = 0; i < 2; ++i) {
      int p = t + i * 256;
      int c = p >> 3;
      int kg = (p & 7) * 4;
      int nt = c >> 4, n = c & 15;
      int hf = (kg >> 3) & 1;
      *(uint2*)(&Bsw[buf][nt][hf * 16 + n][8 * ((kg >> 4) & 1) + (kg & 7)]) = rb[i];
    }
  };

  v8f acc0[4], acc1[4];
#pragma unroll
  for (int i = 0; i < 4; ++i)
#pragma unroll
    for (int e = 0; e < 8; ++e) { acc0[i][e] = 0.f; acc1[i][e] = 0.f; }

  gload(0);
  sstore(0);
  __syncthreads();

  int cur = 0;
  for (int kt = 0; kt < Kt; kt += 32) {
    const bool pref = (kt + 32) < Kt;
    if (pref) gload(kt + 32);      // overlaps with WMMAs below

    v16bf a0 = *(const v16bf*)(&Asw[cur][2 * w][lane][0]);
    v16bf a1 = *(const v16bf*)(&Asw[cur][2 * w + 1][lane][0]);
#pragma unroll
    for (int nt = 0; nt < 4; ++nt) {
      v16bf b = *(const v16bf*)(&Bsw[cur][nt][lane][0]);
      acc0[nt] = __builtin_amdgcn_wmma_f32_16x16x32_bf16(false, a0, false, b,
                                                         (short)0, acc0[nt], false, false);
      acc1[nt] = __builtin_amdgcn_wmma_f32_16x16x32_bf16(false, a1, false, b,
                                                         (short)0, acc1[nt], false, false);
    }
    if (pref) sstore(cur ^ 1);     // other buffer: no WAR hazard
    __syncthreads();               // single barrier per K-step
    cur ^= 1;
  }

#pragma unroll
  for (int st = 0; st < 2; ++st) {
    const v8f* acc = st ? acc1 : acc0;
#pragma unroll
    for (int nt = 0; nt < 4; ++nt)
#pragma unroll
      for (int r = 0; r < 8; ++r) {
        long gr = row0 + w * 32 + st * 16 + r + 8 * half;
        long gc = col0 + nt * 16 + m;
        float v = acc[nt][r];
        if (OUT_F32) {
          if (bias) v += bias[gc];
          ((float*)Cp)[gr * N + gc] = v;
        } else {
          ((__bf16*)Cp)[gr * N + gc] = (__bf16)v;
        }
      }
  }
}

// ---------------------------------------------------------------------------
// Fused attention, all-bf16 I/O: per block = one (b, h, 64-row S tile).
// 77 keys padded to 96 (3 WMMA K-steps). Softmax in f32.
// Block: 128 threads = 4 waves; each wave owns 16 query rows.
// ---------------------------------------------------------------------------
#define ATT_KPAD 96
__global__ __launch_bounds__(128) void attn_wmma(const __bf16* __restrict__ Q,
                                                 const __bf16* __restrict__ Kp,
                                                 const __bf16* __restrict__ Vp,
                                                 __bf16* __restrict__ O,
                                                 float scale) {
  const int NKEY = 77, DH = 64, SEQ = 4096, HD = 1024;
  __shared__ __align__(32) __bf16 Kl[ATT_KPAD][64];        // [key][dh] 12 KB
  __shared__ __align__(32) __bf16 Vt[64][ATT_KPAD];        // [dh][key] 12 KB
  __shared__ float  Sc[4][16][ATT_KPAD];                   // 24 KB
  __shared__ __align__(32) __bf16 Pr[4][16][ATT_KPAD];     // 12 KB

  const int t = threadIdx.x;
  const int lane = t & 31;
  const int w = t >> 5;
  const int half = (lane >> 4) & 1;
  const int m = lane & 15;
  const int s0 = blockIdx.x * 64;
  const int h = blockIdx.y;
  const int b = blockIdx.z;

  // Stage K (row-major, raw 8B copies) and V (transposed); keys 77..95 zero.
  for (int idx = t; idx < ATT_KPAD * 16; idx += 128) {  // groups of 4 bf16
    int r = idx >> 4;
    int c = (idx & 15) * 4;
    uint2 kv = {0u, 0u}, vv = {0u, 0u};
    if (r < NKEY) {
      long g = ((long)b * NKEY + r) * HD + h * DH + c;
      kv = *(const uint2*)(Kp + g);
      vv = *(const uint2*)(Vp + g);
    }
    *(uint2*)(&Kl[r][c]) = kv;
    const __bf16* vh = (const __bf16*)&vv;
    Vt[c + 0][r] = vh[0];
    Vt[c + 1][r] = vh[1];
    Vt[c + 2][r] = vh[2];
    Vt[c + 3][r] = vh[3];
  }
  __syncthreads();

  // Q fragments: raw 16B chunks (slot == offset within chunk).
  const __bf16* qrow = Q + ((long)b * SEQ + s0 + w * 16 + m) * HD + h * DH;
  v8bf h0 = *(const v8bf*)(qrow + half * 16);
  v8bf h1 = *(const v8bf*)(qrow + 8 + half * 16);
  v8bf h2 = *(const v8bf*)(qrow + 32 + half * 16);
  v8bf h3 = *(const v8bf*)(qrow + 40 + half * 16);
  v16bf a0 = __builtin_shufflevector(h0, h1, 0, 1, 2, 3, 4, 5, 6, 7,
                                     8, 9, 10, 11, 12, 13, 14, 15);
  v16bf a1 = __builtin_shufflevector(h2, h3, 0, 1, 2, 3, 4, 5, 6, 7,
                                     8, 9, 10, 11, 12, 13, 14, 15);

  // scores = (Q K^T) * scale. B-frag = contiguous 32B slice of a K row.
#pragma unroll
  for (int nt = 0; nt < 6; ++nt) {
    v8f s;
#pragma unroll
    for (int e = 0; e < 8; ++e) s[e] = 0.f;
    const __bf16* krow = &Kl[nt * 16 + m][0];
    v16bf b0 = *(const v16bf*)(krow + half * 16);
    v16bf b1 = *(const v16bf*)(krow + 32 + half * 16);
    s = __builtin_amdgcn_wmma_f32_16x16x32_bf16(false, a0, false, b0, (short)0, s, false, false);
    s = __builtin_amdgcn_wmma_f32_16x16x32_bf16(false, a1, false, b1, (short)0, s, false, false);
#pragma unroll
    for (int r = 0; r < 8; ++r)
      Sc[w][r + 8 * half][nt * 16 + m] = s[r] * scale;
  }
  __syncthreads();

  // Row softmax over 77 keys; padded columns -> prob 0. Packed b32 stores.
  if (lane < 16) {
    float* row = &Sc[w][m][0];
    float mx = -1e30f;
    for (int c = 0; c < NKEY; ++c) mx = fmaxf(mx, row[c]);
    float sum = 0.f;
    for (int c = 0; c < NKEY; ++c) { float e = __expf(row[c] - mx); row[c] = e; sum += e; }
    float inv = 1.f / sum;
    for (int c = 0; c < ATT_KPAD; c += 2) {
      float p0 = (c < NKEY)     ? row[c] * inv     : 0.f;
      float p1 = (c + 1 < NKEY) ? row[c + 1] * inv : 0.f;
      *(unsigned*)(&Pr[w][m][c]) = pk2(p0, p1);
    }
  }
  __syncthreads();

  // out = P @ V : reduction over 96 keys (3 K-steps), N over Dh (4 tiles).
  v8f o[4];
#pragma unroll
  for (int i = 0; i < 4; ++i)
#pragma unroll
    for (int e = 0; e < 8; ++e) o[i][e] = 0.f;

  const __bf16* prow = &Pr[w][m][0];
#pragma unroll
  for (int ks = 0; ks < ATT_KPAD; ks += 32) {
    v8bf c0 = *(const v8bf*)(prow + ks + half * 16);
    v8bf c1 = *(const v8bf*)(prow + ks + 8 + half * 16);
    v16bf pa = __builtin_shufflevector(c0, c1, 0, 1, 2, 3, 4, 5, 6, 7,
                                       8, 9, 10, 11, 12, 13, 14, 15);
#pragma unroll
    for (int nt = 0; nt < 4; ++nt) {
      v16bf vb = *(const v16bf*)(&Vt[nt * 16 + m][ks + half * 16]);
      o[nt] = __builtin_amdgcn_wmma_f32_16x16x32_bf16(false, pa, false, vb,
                                                      (short)0, o[nt], false, false);
    }
  }
#pragma unroll
  for (int nt = 0; nt < 4; ++nt)
#pragma unroll
    for (int r = 0; r < 8; ++r) {
      long gr = (long)b * SEQ + s0 + w * 16 + r + 8 * half;
      O[gr * HD + h * DH + nt * 16 + m] = (__bf16)o[nt][r];
    }
}

// ---------------------------------------------------------------------------
extern "C" void kernel_launch(void* const* d_in, const int* in_sizes, int n_in,
                              void* d_out, int out_size, void* d_ws, size_t ws_size,
                              hipStream_t stream) {
  const float* X  = (const float*)d_in[0];   // [16,4096,1024]
  const float* E  = (const float*)d_in[1];   // [16,77,768]
  const float* Wq = (const float*)d_in[2];   // [1024,1024]
  const float* Wk = (const float*)d_in[3];   // [768,1024]
  const float* Wv = (const float*)d_in[4];   // [768,1024]
  const float* Wo = (const float*)d_in[5];   // [1024,1024]
  const float* bo = (const float*)d_in[6];   // [1024]
  float* out = (float*)d_out;

  const int Bb = 16, S = 4096, D = 1024, NK = 77, DE = 768, INNER = 1024;
  const long MQ   = (long)Bb * S;    // 65536 (multiple of 256)
  const long MKV  = (long)Bb * NK;   // 1232 valid rows
  const long MKVP = 1280;            // padded to multiple of 256

  // Workspace (bf16): Xb | Qb | Ab | Eb | Kb | Vb | WqT | WkT | WvT | WoT
  __bf16* Xb  = (__bf16*)d_ws;
  __bf16* Qb  = Xb  + MQ * INNER;
  __bf16* Ab  = Qb  + MQ * INNER;
  __bf16* Eb  = Ab  + MQ * INNER;
  __bf16* Kb  = Eb  + MKVP * DE;
  __bf16* Vb  = Kb  + MKVP * INNER;
  __bf16* WqT = Vb  + MKVP * INNER;
  __bf16* WkT = WqT + (long)D * INNER;
  __bf16* WvT = WkT + (long)DE * INNER;
  __bf16* WoT = WvT + (long)DE * INNER;

  dim3 blk(256);
  // prep: bf16 conversions + transposed bf16 weights
  convx<<<dim3((unsigned)(MQ * INNER / 1024)), blk, 0, stream>>>(X, Xb);
  convpad<<<dim3((unsigned)(MKVP * DE / 1024)), blk, 0, stream>>>(E, Eb, (int)MKV, DE);
  wtrans<<<dim3(INNER / 32, D / 32), blk, 0, stream>>>(Wq, WqT, D, INNER);
  wtrans<<<dim3(INNER / 32, DE / 32), blk, 0, stream>>>(Wk, WkT, DE, INNER);
  wtrans<<<dim3(INNER / 32, DE / 32), blk, 0, stream>>>(Wv, WvT, DE, INNER);
  wtrans<<<dim3(D / 32, INNER / 32), blk, 0, stream>>>(Wo, WoT, INNER, D);
  // projections (all through the clean all-bf16 fast path)
  gemm_bb<false><<<dim3(INNER / 64, (unsigned)(MQ / 256)), blk, 0, stream>>>(
      Xb, WqT, nullptr, Qb, (int)MQ, INNER, D);
  gemm_bb<false><<<dim3(INNER / 64, (unsigned)(MKVP / 256)), blk, 0, stream>>>(
      Eb, WkT, nullptr, Kb, (int)MKVP, INNER, DE);
  gemm_bb<false><<<dim3(INNER / 64, (unsigned)(MKVP / 256)), blk, 0, stream>>>(
      Eb, WvT, nullptr, Vb, (int)MKVP, INNER, DE);
  // fused softmax attention -> bf16 [B,S,H*Dh]
  attn_wmma<<<dim3(S / 64, 16, Bb), dim3(128), 0, stream>>>(Qb, Kb, Vb, Ab, 0.125f);
  // out = Ab @ Wo + bo (f32 output)
  gemm_bb<true><<<dim3(D / 64, (unsigned)(MQ / 256)), blk, 0, stream>>>(
      Ab, WoT, bo, out, (int)MQ, D, INNER);
}
